// DAHH_29910152249763
// MI455X (gfx1250) — matrix operand order repaired and checked by reading, not compiled, not run on hardware
//
#include <hip/hip_runtime.h>
#include <cstddef>

// ---------------------------------------------------------------------------
// CDNA5 (gfx1250) wave32 WMMA bf16 GEMM + fused epilogues for the DAHH net.
// Register-double-buffered pipeline: global->regs overlaps WMMA from LDS.
// Scatter staging packs k-pairs into b32 LDS stores.
// ---------------------------------------------------------------------------

typedef __attribute__((ext_vector_type(16))) __bf16 v16bf;
typedef __attribute__((ext_vector_type(8)))  __bf16 v8bf;
typedef __attribute__((ext_vector_type(2)))  __bf16 v2bf;
typedef __attribute__((ext_vector_type(8)))  float  v8f;

#define BM 128
#define BN 128
#define BK 32
#define LDPAD 8   // bf16 elements of row padding; keeps rows 16B aligned (80B rows)

enum { MODE_PLAIN = 0, MODE_ROWSCALE = 1, MODE_CONV = 2, MODE_TANH = 3, MODE_BIAS = 4 };

__device__ __forceinline__ void cvt16(const float4* p, v8bf& lo, v8bf& hi)
{
    float4 f0 = p[0], f1 = p[1], f2 = p[2], f3 = p[3];
    lo[0] = (__bf16)f0.x; lo[1] = (__bf16)f0.y; lo[2] = (__bf16)f0.z; lo[3] = (__bf16)f0.w;
    lo[4] = (__bf16)f1.x; lo[5] = (__bf16)f1.y; lo[6] = (__bf16)f1.z; lo[7] = (__bf16)f1.w;
    hi[0] = (__bf16)f2.x; hi[1] = (__bf16)f2.y; hi[2] = (__bf16)f2.z; hi[3] = (__bf16)f2.w;
    hi[4] = (__bf16)f3.x; hi[5] = (__bf16)f3.y; hi[6] = (__bf16)f3.z; hi[7] = (__bf16)f3.w;
}

// 16 consecutive elements of one row (minor-dim contiguous)
__device__ __forceinline__ void load_strip16(const float* __restrict__ P, int ld,
                                             int grow, int gcol, int nrows, int ncols,
                                             bool fast, v8bf& lo, v8bf& hi)
{
    if (fast) {
        cvt16((const float4*)(P + (size_t)grow * ld + gcol), lo, hi);
    } else {
#pragma unroll
        for (int i = 0; i < 8; ++i) {
            int c0 = gcol + i, c1 = gcol + 8 + i;
            bool rowOK = grow < nrows;
            lo[i] = (__bf16)((rowOK && c0 < ncols) ? P[(size_t)grow * ld + c0] : 0.0f);
            hi[i] = (__bf16)((rowOK && c1 < ncols) ? P[(size_t)grow * ld + c1] : 0.0f);
        }
    }
}

// 8 consecutive columns of TWO adjacent rows (for k-pair packed scatter staging)
__device__ __forceinline__ void load_pair8(const float* __restrict__ P, int ld,
                                           int grow, int gcol, int nrows, int ncols,
                                           bool fast, v8bf& r0, v8bf& r1)
{
    if (fast) {
        const float4* p0 = (const float4*)(P + (size_t)grow * ld + gcol);
        const float4* p1 = (const float4*)(P + (size_t)(grow + 1) * ld + gcol);
        float4 f0 = p0[0], f1 = p0[1], f2 = p1[0], f3 = p1[1];
        r0[0] = (__bf16)f0.x; r0[1] = (__bf16)f0.y; r0[2] = (__bf16)f0.z; r0[3] = (__bf16)f0.w;
        r0[4] = (__bf16)f1.x; r0[5] = (__bf16)f1.y; r0[6] = (__bf16)f1.z; r0[7] = (__bf16)f1.w;
        r1[0] = (__bf16)f2.x; r1[1] = (__bf16)f2.y; r1[2] = (__bf16)f2.z; r1[3] = (__bf16)f2.w;
        r1[4] = (__bf16)f3.x; r1[5] = (__bf16)f3.y; r1[6] = (__bf16)f3.z; r1[7] = (__bf16)f3.w;
    } else {
#pragma unroll
        for (int i = 0; i < 8; ++i) {
            int c = gcol + i;
            bool cOK = c < ncols;
            r0[i] = (__bf16)(((grow     < nrows) && cOK) ? P[(size_t)grow * ld + c]       : 0.0f);
            r1[i] = (__bf16)(((grow + 1 < nrows) && cOK) ? P[(size_t)(grow + 1) * ld + c] : 0.0f);
        }
    }
}

__global__ __launch_bounds__(256)
void gemm_wmma_bf16(const float* __restrict__ A, const float* __restrict__ B,
                    float* __restrict__ C,
                    int M, int K, int Ncol,
                    int lda, int ldb, int ldc,
                    int transA, int mode,
                    const float* __restrict__ rowscale,
                    const float* __restrict__ colbias,
                    const float* __restrict__ resid,
                    float negslope)
{
    __shared__ __bf16 As[BM][BK + LDPAD];   // A tile, [m][k]
    __shared__ __bf16 Bs[BN][BK + LDPAD];   // B tile, transposed: [n][k]

    const int tid  = threadIdx.x;
    const int lane = tid & 31;
    const int wave = tid >> 5;       // 0..7
    const int wm   = wave >> 1;      // 0..3  (wave M position, 32 rows each)
    const int wn   = wave & 1;       // 0..1  (wave N position, 64 cols each)
    const int lrow = lane & 15;
    const int lhi  = lane >> 4;      // 0 or 1 (half-wave)

    const int m0 = blockIdx.y * BM;
    const int n0 = blockIdx.x * BN;

    const bool kMult = (K & (BK - 1)) == 0;
    // uniform-per-block fast flags: branch-free float4 staging when whole side in-bounds
    const bool fastA  = kMult && ((lda & 3) == 0) && (m0 + BM <= M);  // used by both A variants
    const bool fastB  = kMult && ((ldb & 3) == 0) && (n0 + BN <= Ncol);

    // per-thread staging coordinates
    const int arow = tid >> 1, aks = (tid & 1) * 16;     // no-trans A: row, 16-wide k-strip
    const int sp   = (tid & 15) * 8;                     // scatter: 8-wide column strip
    const int skp  = tid >> 4;                           // scatter: k-pair index (0..15)

    v8bf a0, a1, b0, b1;

    // ---- prologue: load k-tile 0 into registers ----
    if (!transA)
        load_strip16(A, lda, m0 + arow, aks, M, K,
                     fastA && (aks + 16 <= K), a0, a1);
    else
        load_pair8(A, lda, 2 * skp, m0 + sp, K, M, fastA && (2 * skp + 2 <= K), a0, a1);
    load_pair8(B, ldb, 2 * skp, n0 + sp, K, Ncol, fastB, b0, b1);

    v8f acc[2][4] = {};

    for (int k0 = 0; k0 < K; k0 += BK) {
        __syncthreads();
        // ---- store current registers to LDS (bf16) ----
        if (!transA) {
            *(v8bf*)&As[arow][aks]     = a0;
            *(v8bf*)&As[arow][aks + 8] = a1;
        } else {
#pragma unroll
            for (int mm = 0; mm < 8; ++mm) {
                v2bf t; t[0] = a0[mm]; t[1] = a1[mm];
                *(v2bf*)&As[sp + mm][2 * skp] = t;    // k-pair packed b32 store
            }
        }
#pragma unroll
        for (int nn = 0; nn < 8; ++nn) {
            v2bf t; t[0] = b0[nn]; t[1] = b1[nn];
            *(v2bf*)&Bs[sp + nn][2 * skp] = t;        // k-pair packed b32 store
        }
        __syncthreads();

        // ---- prefetch next k-tile into registers (overlaps with WMMA below) ----
        int k1 = k0 + BK;
        if (k1 < K) {
            if (!transA)
                load_strip16(A, lda, m0 + arow, k1 + aks, M, K, fastA, a0, a1);
            else
                load_pair8(A, lda, k1 + 2 * skp, m0 + sp, K, M, fastA, a0, a1);
            load_pair8(B, ldb, k1 + 2 * skp, n0 + sp, K, Ncol, fastB, b0, b1);
        }

        // ---- fragments per ISA VGPR layouts (two b128 loads each) ----
        v16bf afrag[2], bfrag[4];
#pragma unroll
        for (int mi = 0; mi < 2; ++mi) {
            int row = wm * 32 + mi * 16 + lrow;
            int kb  = lhi * 8;                       // K base 0 or 8
            v8bf lo = *(const v8bf*)&As[row][kb];       // K = kb..kb+7
            v8bf hi = *(const v8bf*)&As[row][16 + kb];  // K = 16+kb..16+kb+7
            afrag[mi] = __builtin_shufflevector(lo, hi,
                0,1,2,3,4,5,6,7,8,9,10,11,12,13,14,15);
        }
#pragma unroll
        for (int ni = 0; ni < 4; ++ni) {
            int col = wn * 64 + ni * 16 + lrow;
            int kb2 = lhi * 8;                       // K base 0 or 8
            v8bf lo = *(const v8bf*)&Bs[col][kb2];
            v8bf hi = *(const v8bf*)&Bs[col][16 + kb2];
            bfrag[ni] = __builtin_shufflevector(lo, hi,
                0,1,2,3,4,5,6,7,8,9,10,11,12,13,14,15);
        }

        // ---- 8 WMMAs: 32x64 per wave per K-step ----
#pragma unroll
        for (int mi = 0; mi < 2; ++mi)
#pragma unroll
            for (int ni = 0; ni < 4; ++ni)
                acc[mi][ni] = __builtin_amdgcn_wmma_f32_16x16x32_bf16(
                    false, afrag[mi], false, bfrag[ni],
                    (short)0, acc[mi][ni], false, false);
    }

    // ---- epilogue: C layout lane l: N = l&15, VGPR i: M = i + (l>=16 ? 8 : 0) ----
#pragma unroll
    for (int mi = 0; mi < 2; ++mi) {
#pragma unroll
        for (int ni = 0; ni < 4; ++ni) {
            int c = n0 + wn * 64 + ni * 16 + lrow;
            if (c >= Ncol) continue;
            int rbase = m0 + wm * 32 + mi * 16 + lhi * 8;
#pragma unroll
            for (int i = 0; i < 8; ++i) {
                int r = rbase + i;
                if (r >= M) continue;
                float v = acc[mi][ni][i];
                if (mode == MODE_ROWSCALE) {
                    v = v / rowscale[r];
                } else if (mode == MODE_CONV) {
                    v = v / rowscale[r] + colbias[c];
                    v = (v >= 0.0f) ? v : v * negslope;     // leaky relu
                    v += resid[(size_t)r * ldc + c];        // residual X_i
                } else if (mode == MODE_TANH) {
                    v = tanhf(v + colbias[c]);
                } else if (mode == MODE_BIAS) {
                    v = v + colbias[c];
                }
                C[(size_t)r * ldc + c] = v;
            }
        }
    }
}

// ---- stage 1: partial column sums of H over a chunk of rows ----
__global__ void col_sum_part_kernel(const float* __restrict__ H, float* __restrict__ part,
                                    int N, int E, int nchunks)
{
    int e = blockIdx.x * blockDim.x + threadIdx.x;
    int chunk = blockIdx.y;
    if (e >= E) return;
    int per = (N + nchunks - 1) / nchunks;
    int nb = chunk * per;
    int ne = (nb + per < N) ? nb + per : N;
    float s = 0.0f;
    for (int n = nb; n < ne; ++n) s += H[(size_t)n * E + e];
    part[(size_t)chunk * E + e] = s;
}

__global__ void col_sum_finish_kernel(const float* __restrict__ part, float* __restrict__ out,
                                      int E, int nchunks)
{
    int e = blockIdx.x * blockDim.x + threadIdx.x;
    if (e >= E) return;
    float s = 0.0f;
    for (int c = 0; c < nchunks; ++c) s += part[(size_t)c * E + e];
    out[e] = s;
}

// ---- node degrees: one block per row, reduce over E ----
__global__ void row_sum_kernel(const float* __restrict__ H, float* __restrict__ dv,
                               int N, int E)
{
    __shared__ float red[256];
    int n = blockIdx.x;
    float s = 0.0f;
    for (int e = threadIdx.x; e < E; e += blockDim.x) s += H[(size_t)n * E + e];
    red[threadIdx.x] = s;
    __syncthreads();
    for (int off = 128; off > 0; off >>= 1) {
        if ((int)threadIdx.x < off) red[threadIdx.x] += red[threadIdx.x + off];
        __syncthreads();
    }
    if (threadIdx.x == 0) dv[n] = red[0];
}

// ---- column mean of Gamma: out[c] = (1/rows) * sum_r G[r][c] ----
__global__ void col_mean_kernel(const float* __restrict__ G, float* __restrict__ out,
                                int rows, int cols)
{
    int c = blockIdx.x * blockDim.x + threadIdx.x;
    if (c >= cols) return;
    float s = 0.0f;
    for (int r = 0; r < rows; ++r) s += G[(size_t)r * cols + c];
    out[c] = s / (float)rows;
}

// ---- stage 1: partial S/P over a chunk of nodes ----
__global__ void wcs2_part_kernel(const float* __restrict__ Xn,
                                 const float* __restrict__ w2,
                                 const float* __restrict__ w1,
                                 float* __restrict__ Sp, float* __restrict__ Pp,
                                 int N, int D, int nchunks)
{
    int d = blockIdx.x * blockDim.x + threadIdx.x;
    int chunk = blockIdx.y;
    if (d >= D) return;
    int per = (N + nchunks - 1) / nchunks;
    int nb = chunk * per;
    int ne = (nb + per < N) ? nb + per : N;
    float s = 0.0f, p = 0.0f;
    for (int n = nb; n < ne; ++n) {
        float x = Xn[(size_t)n * D + d];
        s += w2[n] * x;
        p += w1[n] * x;
    }
    Sp[(size_t)chunk * D + d] = s;
    Pp[(size_t)chunk * D + d] = p;
}

__global__ void wcs2_finish_kernel(const float* __restrict__ Sp, const float* __restrict__ Pp,
                                   float* __restrict__ S, float* __restrict__ P,
                                   int D, int nchunks)
{
    int d = blockIdx.x * blockDim.x + threadIdx.x;
    if (d >= D) return;
    float s = 0.0f, p = 0.0f;
    for (int c = 0; c < nchunks; ++c) {
        s += Sp[(size_t)c * D + d];
        p += Pp[(size_t)c * D + d];
    }
    S[d] = s;
    P[d] = p;
}

// ---- cb[h] = b[h] + sum_d vec[d] * W[(D+d)][h]  (fold broadcast half of concat) ----
__global__ void combined_bias_kernel(const float* __restrict__ W,
                                     const float* __restrict__ vec,
                                     const float* __restrict__ b,
                                     float* __restrict__ cb, int D, int HID)
{
    int h = blockIdx.x * blockDim.x + threadIdx.x;
    if (h >= HID) return;
    float s = b[h];
    for (int d = 0; d < D; ++d) s += vec[d] * W[(size_t)(D + d) * HID + h];
    cb[h] = s;
}

extern "C" void kernel_launch(void* const* d_in, const int* in_sizes, int n_in,
                              void* d_out, int out_size, void* d_ws, size_t ws_size,
                              hipStream_t stream)
{
    (void)in_sizes; (void)n_in; (void)out_size; (void)ws_size;

    const float* X   = (const float*)d_in[0];   // [N,D]
    const float* H   = (const float*)d_in[1];   // [N,E]
    const float* G2  = (const float*)d_in[2];   // [512,N]
    const float* G1  = (const float*)d_in[3];   // [128,N]
    const float* th1 = (const float*)d_in[4];   // [D,D]
    const float* b1  = (const float*)d_in[5];   // [D]
    const float* th2 = (const float*)d_in[6];   // [D,D]
    const float* b2  = (const float*)d_in[7];   // [D]
    const float* W1a = (const float*)d_in[8];   // [2D,HID]
    const float* b1a = (const float*)d_in[9];   // [HID]
    const float* W2a = (const float*)d_in[10];  // [HID,N1]
    const float* b2a = (const float*)d_in[11];  // [N1]
    const float* W1b = (const float*)d_in[12];  // [2D,HID]
    const float* b1b = (const float*)d_in[13];  // [HID]
    const float* W2b = (const float*)d_in[14];  // [HID,N2]
    const float* b2b = (const float*)d_in[15];  // [N2]

    constexpr int NN = 20000, EE = 4000, DD = 768, HD = 512, NC1 = 29, NC2 = 159;
    constexpr int CHUNKS = 64;

    // workspace carve-up (floats)
    float* ws     = (float*)d_ws;
    float* de     = ws;                         // [E]
    float* dv     = de + EE;                    // [N]
    float* g2bar  = dv + NN;                    // [N]
    float* g1bar  = g2bar + NN;                 // [N]
    float* Svec   = g1bar + NN;                 // [D]
    float* Pvec   = Svec + DD;                  // [D]
    float* cba    = Pvec + DD;                  // [HID]
    float* cbb    = cba + HD;                   // [HID]
    float* depart = cbb + HD;                   // [CHUNKS,E]
    float* Spart  = depart + (size_t)CHUNKS * EE;   // [CHUNKS,D]
    float* Ppart  = Spart + (size_t)CHUNKS * DD;    // [CHUNKS,D]
    float* xt     = Ppart + (size_t)CHUNKS * DD;    // [N,D] (reused as hidden [N,HID])
    float* yb     = xt + (size_t)NN * DD;       // [E,D]
    float* Xn     = yb + (size_t)EE * DD;       // [N,D]

    float* out1 = (float*)d_out;                // [N,29]
    float* out2 = out1 + (size_t)NN * NC1;      // [N,159]

    auto gemm = [&](const float* A, const float* B, float* C,
                    int M, int K, int Ncol, int lda, int ldb, int ldc,
                    int transA, int mode,
                    const float* rs, const float* cb, const float* rez) {
        dim3 grid((Ncol + BN - 1) / BN, (M + BM - 1) / BM);
        hipLaunchKernelGGL(gemm_wmma_bf16, grid, dim3(256), 0, stream,
                           A, B, C, M, K, Ncol, lda, ldb, ldc, transA, mode,
                           rs, cb, rez, 0.01f);
    };

    // ---- degrees & Gamma row-means ----
    hipLaunchKernelGGL(col_sum_part_kernel, dim3((EE + 255) / 256, CHUNKS), dim3(256),
                       0, stream, H, depart, NN, EE, CHUNKS);
    hipLaunchKernelGGL(col_sum_finish_kernel, dim3((EE + 255) / 256), dim3(256),
                       0, stream, depart, de, EE, CHUNKS);
    hipLaunchKernelGGL(row_sum_kernel, dim3(NN), dim3(256), 0, stream, H, dv, NN, EE);
    hipLaunchKernelGGL(col_mean_kernel, dim3((NN + 255) / 256), dim3(256), 0, stream,
                       G2, g2bar, 512, NN);
    hipLaunchKernelGGL(col_mean_kernel, dim3((NN + 255) / 256), dim3(256), 0, stream,
                       G1, g1bar, 128, NN);

    // ---- layer 1: xt = X@theta1 ; y = (H^T@xt)/d_e ; Xn = lrelu(H@y/d_v + b1)+X ----
    gemm(X,  th1, xt, NN, DD, DD, DD, DD, DD, 0, MODE_PLAIN,    nullptr, nullptr, nullptr);
    gemm(H,  xt,  yb, EE, NN, DD, EE, DD, DD, 1, MODE_ROWSCALE, de,      nullptr, nullptr);
    gemm(H,  yb,  Xn, NN, EE, DD, EE, DD, DD, 0, MODE_CONV,     dv,      b1,      X);

    // ---- layer 2 ----
    gemm(Xn, th2, xt, NN, DD, DD, DD, DD, DD, 0, MODE_PLAIN,    nullptr, nullptr, nullptr);
    gemm(H,  xt,  yb, EE, NN, DD, EE, DD, DD, 1, MODE_ROWSCALE, de,      nullptr, nullptr);
    gemm(H,  yb,  Xn, NN, EE, DD, EE, DD, DD, 0, MODE_CONV,     dv,      b2,      X);

    // ---- S = g2bar@Xn, P = g1bar@Xn (two-stage) ; fold broadcast halves into biases ----
    hipLaunchKernelGGL(wcs2_part_kernel, dim3((DD + 255) / 256, CHUNKS), dim3(256),
                       0, stream, Xn, g2bar, g1bar, Spart, Ppart, NN, DD, CHUNKS);
    hipLaunchKernelGGL(wcs2_finish_kernel, dim3((DD + 255) / 256), dim3(256),
                       0, stream, Spart, Ppart, Svec, Pvec, DD, CHUNKS);
    hipLaunchKernelGGL(combined_bias_kernel, dim3((HD + 255) / 256), dim3(256), 0,
                       stream, W1a, Pvec, b1a, cba, DD, HD);
    hipLaunchKernelGGL(combined_bias_kernel, dim3((HD + 255) / 256), dim3(256), 0,
                       stream, W1b, Svec, b1b, cbb, DD, HD);

    // ---- MLP 1: hid = tanh(Xn@W1a[:D] + cba) ; out1 = hid@W2a + b2a ----
    gemm(Xn, W1a, xt,   NN, DD, HD,  DD, HD,  HD,  0, MODE_TANH, nullptr, cba, nullptr);
    gemm(xt, W2a, out1, NN, HD, NC1, HD, NC1, NC1, 0, MODE_BIAS, nullptr, b2a, nullptr);

    // ---- MLP 2: hid = tanh(Xn@W1b[:D] + cbb) ; out2 = hid@W2b + b2b ----
    gemm(Xn, W1b, xt,   NN, DD, HD,  DD, HD,  HD,  0, MODE_TANH, nullptr, cbb, nullptr);
    gemm(xt, W2b, out2, NN, HD, NC2, HD, NC2, NC2, 0, MODE_BIAS, nullptr, b2b, nullptr);
}
